// MoEScoreHead_26482768347645
// MI455X (gfx1250) — matrix operand order repaired and checked by reading, not compiled, not run on hardware
//
#include <hip/hip_runtime.h>

#define B_TOK 16384
#define H_DIM 2048
#define E_NUM 8
#define HF_DIM 512
#define TM 128
#define TF 128
#define KSTEP 32
#define LDA 40   // padded LDS row stride (fp16 elems) -> 80B, conflict-free frag loads

typedef __attribute__((ext_vector_type(16))) _Float16 v16h;
typedef __attribute__((ext_vector_type(8)))  float    v8f;

union Frag16 { v16h v; uint4 q[2]; };

// ---------------------------------------------------------------------------
// Kernel 0: zero the per-expert routing counters (ws is not re-poisoned by the
// harness, so we must reset them deterministically every launch).
// ---------------------------------------------------------------------------
__global__ void moe_zero_counts(int* __restrict__ counts) {
    if (threadIdx.x < E_NUM) counts[threadIdx.x] = 0;
}

// ---------------------------------------------------------------------------
// Kernel 1: gating. One wave32 per token: logits = x[b] @ Wg + bg, top-2,
// softmax over the top-2, write logits to d_out, zero final_scores slot,
// and append (token, weight) to the two chosen experts' dispatch lists.
// ---------------------------------------------------------------------------
__global__ void moe_gate(const float* __restrict__ x, const float* __restrict__ Wg,
                         const float* __restrict__ bg, float* __restrict__ out,
                         int* __restrict__ counts, int* __restrict__ lists,
                         float* __restrict__ wlist) {
    const int wave = threadIdx.x >> 5;
    const int lane = threadIdx.x & 31;
    const int tok  = blockIdx.x * 8 + wave;
    const float* xr = x + (size_t)tok * H_DIM;

    float acc[E_NUM];
#pragma unroll
    for (int e = 0; e < E_NUM; ++e) acc[e] = 0.0f;

    for (int i = 0; i < H_DIM / 32; ++i) {
        const int h = i * 32 + lane;
        const float xv = xr[h];                       // coalesced across lanes
        const float* wr = Wg + (size_t)h * E_NUM;     // 32B-aligned row of 8
        const float4 wA = *(const float4*)(wr);
        const float4 wB = *(const float4*)(wr + 4);
        acc[0] = fmaf(xv, wA.x, acc[0]); acc[1] = fmaf(xv, wA.y, acc[1]);
        acc[2] = fmaf(xv, wA.z, acc[2]); acc[3] = fmaf(xv, wA.w, acc[3]);
        acc[4] = fmaf(xv, wB.x, acc[4]); acc[5] = fmaf(xv, wB.y, acc[5]);
        acc[6] = fmaf(xv, wB.z, acc[6]); acc[7] = fmaf(xv, wB.w, acc[7]);
    }
    // full-wave butterfly reduce: afterwards every lane holds all 8 logits
#pragma unroll
    for (int e = 0; e < E_NUM; ++e) {
        float v = acc[e];
#pragma unroll
        for (int off = 16; off >= 1; off >>= 1) v += __shfl_xor(v, off, 32);
        acc[e] = v + bg[e];
    }

    if (lane < E_NUM)   // logits output region
        out[B_TOK + (size_t)tok * E_NUM + lane] = acc[lane];

    if (lane == 0) {
        out[tok] = 0.0f;   // zero final_scores slot (accumulated by kernel 2)
        // top-2, first-index tie-break like lax.top_k
        int i0 = 0;
#pragma unroll
        for (int e = 1; e < E_NUM; ++e) if (acc[e] > acc[i0]) i0 = e;
        int i1 = (i0 == 0) ? 1 : 0;
#pragma unroll
        for (int e = 0; e < E_NUM; ++e) {
            if (e == i0) continue;
            if (acc[e] > acc[i1]) i1 = e;
        }
        const float ex = expf(acc[i1] - acc[i0]);   // <= 1, numerically stable
        const float w0 = 1.0f / (1.0f + ex);
        const float w1 = ex   / (1.0f + ex);
        const int s0 = atomicAdd(&counts[i0], 1);
        lists[i0 * B_TOK + s0] = tok; wlist[i0 * B_TOK + s0] = w0;
        const int s1 = atomicAdd(&counts[i1], 1);
        lists[i1 * B_TOK + s1] = tok; wlist[i1 * B_TOK + s1] = w1;
    }
}

// fused exact GELU (matches approximate=False)
__device__ __forceinline__ float gelu_exact(float v) {
    return 0.5f * v * (1.0f + erff(v * 0.70710678118654752f));
}

// ---------------------------------------------------------------------------
// Kernel 2: grouped expert GEMM over routed tokens only.
// Block = 256 threads = 8 waves (2 M x 4 N), tile 128 tokens x 128 HF, K=32.
// Register double-buffering: next K-step's global loads are issued before the
// WMMAs so their latency hides under compute; all addresses hoisted out of
// the loop so no per-iteration LDS reloads of sTok survive the barriers.
// ---------------------------------------------------------------------------
__launch_bounds__(256)
__global__ void moe_expert(const float* __restrict__ x, const float* __restrict__ W1,
                           const float* __restrict__ b1, const float* __restrict__ W2,
                           const int* __restrict__ counts, const int* __restrict__ lists,
                           const float* __restrict__ wlist, float* __restrict__ out) {
    const int e = blockIdx.z;
    const int nTok = counts[e];
    const int base = blockIdx.x * TM;
    if (base >= nTok) return;               // no routed tokens in this chunk
    const int fbase = blockIdx.y * TF;

    __shared__ int      sTok[TM];
    __shared__ float    sW[TM];
    __shared__ _Float16 As[TM * LDA];       // x tile   [token][k]  (fp16)
    __shared__ _Float16 Ws[TF * LDA];       // W1 tile  [f][k]      (transposed)

    const int tid = threadIdx.x;
    if (tid < TM) {
        const int slot = base + tid;
        const int t = (slot < nTok) ? lists[e * B_TOK + slot] : 0;  // dummy row 0 if tail
        sTok[tid] = t;
        sW[tid]   = (slot < nTok) ? wlist[e * B_TOK + slot] : 0.0f;
    }
    __syncthreads();

    // ---- hoist every address out of the K loop (sTok read exactly once) ----
    const float* W1e = W1 + (size_t)e * H_DIM * HF_DIM;
    const int xc4 = (tid & 7) << 2;     // x  col within K-step: 0..28
    const int wf4 = (tid & 31) << 2;    // W1 col within f-tile: 0..124
    const float* xbase[4];
    const float* wbase[4];
    _Float16*    asdst[4];
    _Float16*    wsdst[4];
#pragma unroll
    for (int i = 0; i < 4; ++i) {
        const int xrow = (tid >> 3) + 32 * i;                 // 0..127
        xbase[i] = x + (size_t)sTok[xrow] * H_DIM + xc4;
        asdst[i] = &As[xrow * LDA + xc4];
        const int hh = (tid >> 5) + 8 * i;                    // 0..31
        wbase[i] = W1e + (size_t)hh * HF_DIM + fbase + wf4;
        wsdst[i] = &Ws[wf4 * LDA + hh];                       // +u*LDA, u=0..3
    }

    const int wave = tid >> 5, lane = tid & 31;
    const int wm = wave & 1;        // M half: token rows [wm*64, wm*64+64)
    const int wn = wave >> 1;       // N quarter: f cols  [wn*32, wn*32+32)
    const int lr = lane & 15;       // lane within 16-group
    const int lh = lane >> 4;       // lane half (WMMA K-sub-block select)

    const _Float16* afrag[4];
    const _Float16* bfrag[2];
#pragma unroll
    for (int m = 0; m < 4; ++m)
        afrag[m] = &As[(wm * 64 + m * 16 + lr) * LDA + lh * 8];
#pragma unroll
    for (int nn = 0; nn < 2; ++nn)
        bfrag[nn] = &Ws[(wn * 32 + nn * 16 + lr) * LDA + lh * 8];

    v8f acc[4][2] = {};
    float4 rx[4], rw[4];

    // ---- preload + stage K-step 0 ----
#pragma unroll
    for (int i = 0; i < 4; ++i) {
        rx[i] = *(const float4*)(xbase[i]);
        rw[i] = *(const float4*)(wbase[i]);
    }
#pragma unroll
    for (int i = 0; i < 4; ++i) {
        _Float16* d = asdst[i];
        d[0] = (_Float16)rx[i].x; d[1] = (_Float16)rx[i].y;
        d[2] = (_Float16)rx[i].z; d[3] = (_Float16)rx[i].w;
        _Float16* w = wsdst[i];
        w[0]       = (_Float16)rw[i].x;
        w[LDA]     = (_Float16)rw[i].y;
        w[2 * LDA] = (_Float16)rw[i].z;
        w[3 * LDA] = (_Float16)rw[i].w;
    }
    __syncthreads();

    for (int kk = 0; kk < H_DIM; kk += KSTEP) {
        const bool notLast = (kk + KSTEP) < H_DIM;
        if (notLast) {
            // issue next tile's global loads; latency hides under the WMMAs
#pragma unroll
            for (int i = 0; i < 4; ++i) {
                rx[i] = *(const float4*)(xbase[i] + (kk + KSTEP));
                rw[i] = *(const float4*)(wbase[i] + (size_t)(kk + KSTEP) * HF_DIM);
            }
            // branchless speculative prefetch two K-steps ahead (wrapped)
            __builtin_prefetch(wbase[0] + (size_t)((kk + 2 * KSTEP) & (H_DIM - 1)) * HF_DIM, 0, 3);
        }

        // A frags per ISA layout: lane-half lh selects K {lh*8..+7},{16+lh*8..+7}
        Frag16 a[4], b[2];
#pragma unroll
        for (int m = 0; m < 4; ++m) {
            a[m].q[0] = *(const uint4*)(afrag[m]);
            a[m].q[1] = *(const uint4*)(afrag[m] + 16);
        }
        // B frags: lane-half lh selects K {lh*8..+7},{16+lh*8..+7} (same as A)
#pragma unroll
        for (int nn = 0; nn < 2; ++nn) {
            b[nn].q[0] = *(const uint4*)(bfrag[nn]);
            b[nn].q[1] = *(const uint4*)(bfrag[nn] + 16);
        }
#pragma unroll
        for (int m = 0; m < 4; ++m)
#pragma unroll
            for (int nn = 0; nn < 2; ++nn)
                acc[m][nn] = __builtin_amdgcn_wmma_f32_16x16x32_f16(
                    false, a[m].v, false, b[nn].v, (short)0, acc[m][nn], false, false);
        __syncthreads();   // frag reads done -> LDS reusable

        if (notLast) {
#pragma unroll
            for (int i = 0; i < 4; ++i) {
                _Float16* d = asdst[i];
                d[0] = (_Float16)rx[i].x; d[1] = (_Float16)rx[i].y;
                d[2] = (_Float16)rx[i].z; d[3] = (_Float16)rx[i].w;
                _Float16* w = wsdst[i];
                w[0]       = (_Float16)rw[i].x;
                w[LDA]     = (_Float16)rw[i].y;
                w[2 * LDA] = (_Float16)rw[i].z;
                w[3 * LDA] = (_Float16)rw[i].w;
            }
            __syncthreads();
        }
    }

    // Epilogue: +b1, exact GELU, * W2[e][f], reduce over this wave's 32 f-cols,
    // then combine-weighted atomic accumulate into final_scores.
#pragma unroll
    for (int m = 0; m < 4; ++m) {
        float csum[8];
#pragma unroll
        for (int v = 0; v < 8; ++v) csum[v] = 0.0f;
#pragma unroll
        for (int nn = 0; nn < 2; ++nn) {
            const int f  = fbase + wn * 32 + nn * 16 + lr;   // N = lane%16
            const float bb = b1[e * HF_DIM + f];
            const float w2 = W2[(size_t)e * HF_DIM + f];
#pragma unroll
            for (int v = 0; v < 8; ++v)
                csum[v] = fmaf(gelu_exact(acc[m][nn][v] + bb), w2, csum[v]);
        }
#pragma unroll
        for (int v = 0; v < 8; ++v) {
            float c = csum[v];
            c += __shfl_xor(c, 1, 32); c += __shfl_xor(c, 2, 32);
            c += __shfl_xor(c, 4, 32); c += __shfl_xor(c, 8, 32);
            if (lr == 0) {                              // lanes 0 and 16 own sums
                const int slot = wm * 64 + m * 16 + v + 8 * lh;  // C: M = v + 8*lh
                if (base + slot < nTok)
                    atomicAdd(&out[sTok[slot]], sW[slot] * c);
            }
        }
    }
}

extern "C" void kernel_launch(void* const* d_in, const int* in_sizes, int n_in,
                              void* d_out, int out_size, void* d_ws, size_t ws_size,
                              hipStream_t stream) {
    const float* x  = (const float*)d_in[0];
    const float* W1 = (const float*)d_in[1];
    const float* b1 = (const float*)d_in[2];
    const float* W2 = (const float*)d_in[3];
    const float* Wg = (const float*)d_in[4];
    const float* bg = (const float*)d_in[5];
    float* out = (float*)d_out;   // [0,B): final_scores, [B, B+B*E): logits

    char* ws = (char*)d_ws;
    int*   counts = (int*)ws;                                           // 8 ints
    int*   lists  = (int*)(ws + 256);                                   // E*B ints
    float* wlist  = (float*)(ws + 256 + sizeof(int) * E_NUM * B_TOK);   // E*B floats

    moe_zero_counts<<<1, 64, 0, stream>>>(counts);
    moe_gate<<<B_TOK / 8, 256, 0, stream>>>(x, Wg, bg, out, counts, lists, wlist);
    dim3 grid(B_TOK / TM, HF_DIM / TF, E_NUM);
    moe_expert<<<grid, 256, 0, stream>>>(x, W1, b1, W2, counts, lists, wlist, out);
}
